// DeltaRule_22488448762123
// MI455X (gfx1250) — matrix operand order repaired and checked by reading, not compiled, not run on hardware
//
#include <hip/hip_runtime.h>

typedef __bf16 bf16_t;
typedef __attribute__((ext_vector_type(16))) __bf16 v16bf;
typedef __attribute__((ext_vector_type(8)))  float  v8f;

#define T_TOTAL 4096
#define DDIM    128
#define CHUNK   64
#define NCHUNK  (T_TOTAL / CHUNK)
#define BETA_MIN 0.01f
#define BETA_MAX 0.995f
#define EPSV     1e-6f

// Padded LDS strides: odd multiples of 16B -> bank-conflict-free fragment rows,
// still 16B aligned so ds_load_b128 merging works.
#define LDQ 136   // bf16 row-major stride (Qb, Kb, Hb shadows, Zb): 272B = 17*16
#define LDT 72    // bf16 transposed stride (Kts, Vt, Alds):         144B = 9*16
#define LDH 132   // f32 H master stride:                            528B = 33*16

// ---- LDS layout (bytes), every region 16B aligned ----
#define OFF_QB    0u         // 64 x LDQ bf16 : phi(q), row major
#define OFF_KB    17408u     // 64 x LDQ bf16 : phi(k), row major
#define OFF_KTS1  34816u     // 128 x LDT bf16: phi(k)^T * e^{L1C-L1_s} * m_s
#define OFF_KTS2  53248u
#define OFF_VT    71680u     // 128 x LDT bf16: v^T
#define OFF_A     90112u     // 64 x LDT bf16 : combined scaled causal attention
#define OFF_HF1   99328u     // 128 x LDH f32 : H1 master, transposed (Ht[e][d])
#define OFF_HF2   166912u
#define OFF_HB1   234496u    // 128 x LDQ bf16: H1 shadow, transposed
#define OFF_HB2   269312u
#define OFF_Z1    304128u    // 128 f32
#define OFF_Z2    304640u
#define OFF_L1    305152u    // 64 f32 inclusive log-cumsum b1
#define OFF_L2    305408u
#define OFF_E1    305664u    // 64 f32 e^{L1_t}
#define OFF_E2    305920u
#define OFF_X1    306176u    // 64 f32 e^{L1C-L1_t} * m_t
#define OFF_X2    306432u
#define OFF_MV    306688u    // 64 f32 mask
#define OFF_DENA  306944u    // 64 f32 rowsum(A)
#define OFF_DENC1 307200u    // 64 f32 phi(q).Z1
#define OFF_OSC   307456u    // 64 f32 m_t / den
#define OFF_O1    307712u    // 64 f32 E1 * osc
#define OFF_O2    307968u    // 64 f32 E2 * osc
#define OFF_ONES  308224u    // 16 x LDT bf16 all-ones (sum-by-WMMA B operand)
#define OFF_ZB    310528u    // 16 x LDQ bf16 : row0 = Z1, row1 = Z2, rest 0
#define OFF_DENC2 314880u    // 64 f32 phi(q).Z2
#define LDS_BYTES 315136u

__device__ inline v8f wmma_bf16(v16bf a, v16bf b, v8f c) {
  return __builtin_amdgcn_wmma_f32_16x16x32_bf16(false, a, false, b, (short)0, c,
                                                 false, false);
}

// A operand: 16x32 tile, row-major source.
// Lane: row M = lane&15; K pairs {8h+0..7, 16+8h+0..7}, h = lane>>4.
__device__ inline v16bf load_a_bf16(const bf16_t* p, int ld, int row0, int col0, int lane) {
  int m = lane & 15, half = lane >> 4;
  const bf16_t* r0 = p + (row0 + m) * ld + col0 + half * 8;
  const bf16_t* r1 = r0 + 16;
  v16bf a;
#pragma unroll
  for (int i = 0; i < 4; ++i) {
    a[2 * i]     = r0[2 * i];
    a[2 * i + 1] = r0[2 * i + 1];
    a[8 + 2 * i] = r1[2 * i];
    a[9 + 2 * i] = r1[2 * i + 1];
  }
  return a;
}

// B operand: 32x16 tile, element B[k][n] = src[(n0+n)*ld + k0 + k]
// (source stored transposed so k runs contiguously per lane).
__device__ inline v16bf load_b_bf16_t(const bf16_t* p, int ld, int n0, int k0, int lane) {
  int n = lane & 15, kh = (lane >> 4) * 8;
  const bf16_t* r = p + (n0 + n) * ld + k0 + kh;
  v16bf b;
#pragma unroll
  for (int i = 0; i < 8; ++i) { b[i] = r[i]; b[8 + i] = r[16 + i]; }
  return b;
}

__global__ __launch_bounds__(256)
void delta_chunk_kernel(const float* __restrict__ q, const float* __restrict__ k,
                        const float* __restrict__ v, const float* __restrict__ beta,
                        const float* __restrict__ mask,
                        const float* __restrict__ bb1, const float* __restrict__ bb2,
                        float* __restrict__ out) {
  extern __shared__ char smem[];
  bf16_t* Qb   = (bf16_t*)(smem + OFF_QB);
  bf16_t* Kb   = (bf16_t*)(smem + OFF_KB);
  bf16_t* Kts1 = (bf16_t*)(smem + OFF_KTS1);
  bf16_t* Kts2 = (bf16_t*)(smem + OFF_KTS2);
  bf16_t* Vt   = (bf16_t*)(smem + OFF_VT);
  bf16_t* Alds = (bf16_t*)(smem + OFF_A);
  float*  Hf1  = (float*)(smem + OFF_HF1);
  float*  Hf2  = (float*)(smem + OFF_HF2);
  bf16_t* Hb1  = (bf16_t*)(smem + OFF_HB1);
  bf16_t* Hb2  = (bf16_t*)(smem + OFF_HB2);
  float*  Z1   = (float*)(smem + OFF_Z1);
  float*  Z2   = (float*)(smem + OFF_Z2);
  float*  L1   = (float*)(smem + OFF_L1);
  float*  L2   = (float*)(smem + OFF_L2);
  float*  E1   = (float*)(smem + OFF_E1);
  float*  E2   = (float*)(smem + OFF_E2);
  float*  X1   = (float*)(smem + OFF_X1);
  float*  X2   = (float*)(smem + OFF_X2);
  float*  mvals= (float*)(smem + OFF_MV);
  float*  denA = (float*)(smem + OFF_DENA);
  float*  denC1= (float*)(smem + OFF_DENC1);
  float*  denC2= (float*)(smem + OFF_DENC2);
  float*  oscal= (float*)(smem + OFF_OSC);
  float*  O1   = (float*)(smem + OFF_O1);
  float*  O2   = (float*)(smem + OFF_O2);
  bf16_t* OnesB= (bf16_t*)(smem + OFF_ONES);
  bf16_t* Zb   = (bf16_t*)(smem + OFF_ZB);

  const int b    = blockIdx.x;
  const int tid  = threadIdx.x;
  const int lane = tid & 31;
  const int wave = tid >> 5;
  const int n    = lane & 15;
  const int half = lane >> 4;

  const float* qB = q + (size_t)b * T_TOTAL * DDIM;
  const float* kB = k + (size_t)b * T_TOTAL * DDIM;
  const float* vB = v + (size_t)b * T_TOTAL * DDIM;
  const float* bet = beta + (size_t)b * T_TOTAL;
  const float* msk = mask + (size_t)b * T_TOTAL;
  float* outB = out + (size_t)b * T_TOTAL * DDIM;

  const float b1base = fminf(fmaxf(1.f / (1.f + __expf(-bb1[0])), BETA_MIN), BETA_MAX);
  const float b2base = fminf(fmaxf(1.f / (1.f + __expf(-bb2[0])), BETA_MIN), BETA_MAX);

  // zero states / init constants (padded regions included)
  for (int i = tid; i < DDIM * LDH; i += 256) { Hf1[i] = 0.f; Hf2[i] = 0.f; }
  for (int i = tid; i < DDIM * LDQ; i += 256) { Hb1[i] = (bf16_t)0.f; Hb2[i] = (bf16_t)0.f; }
  for (int i = tid; i < 16 * LDT; i += 256) OnesB[i] = (bf16_t)1.f;
  for (int i = tid; i < 16 * LDQ; i += 256) Zb[i] = (bf16_t)0.f;
  if (tid < DDIM) { Z1[tid] = 0.f; Z2[tid] = 0.f; }
  __syncthreads();

  for (int c = 0; c < NCHUNK; ++c) {
    const int t0 = c * CHUNK;

    // ---- phase 0: per-step decays, log space ----
    if (tid < CHUNK) {
      float bi = fminf(fmaxf(bet[t0 + tid], BETA_MIN), BETA_MAX);
      float b1 = fminf(fmaxf(b1base * bi, BETA_MIN), BETA_MAX);
      float b2 = fminf(fmaxf(b2base * bi, BETA_MIN), BETA_MAX);
      L1[tid] = __logf(b1);
      L2[tid] = __logf(b2);
      mvals[tid] = msk[t0 + tid];
    }
    __syncthreads();
    if (tid == 0) {
      float a1 = 0.f, a2 = 0.f;
#pragma unroll 8
      for (int t = 0; t < CHUNK; ++t) {
        a1 += L1[t]; L1[t] = a1;
        a2 += L2[t]; L2[t] = a2;
      }
    }
    __syncthreads();
    const float LC1 = L1[CHUNK - 1], LC2 = L2[CHUNK - 1];
    if (tid < CHUNK) {  // hoisted transcendentals for this chunk
      float l1 = L1[tid], l2 = L2[tid], mt = mvals[tid];
      E1[tid] = __expf(l1);
      E2[tid] = __expf(l2);
      X1[tid] = __expf(LC1 - l1) * mt;
      X2[tid] = __expf(LC2 - l2) * mt;
    }
    __syncthreads();

    // ---- phase 1: load chunk, apply phi, stage LDS ----
#pragma unroll
    for (int j = 0; j < 8; ++j) {
      int idx4 = tid + j * 256;
      int e = idx4 << 2;
      int t = e >> 7;
      int d0 = e & (DDIM - 1);
      size_t g = (size_t)(t0 + t) * DDIM + d0;
      if (c + 1 < NCHUNK) __builtin_prefetch(qB + g + CHUNK * DDIM, 0, 1);
      float s1 = X1[t], s2 = X2[t];
      float4 qv = *(const float4*)(qB + g);
      float4 kv = *(const float4*)(kB + g);
      float4 vv = *(const float4*)(vB + g);
      float qa[4] = {qv.x, qv.y, qv.z, qv.w};
      float ka[4] = {kv.x, kv.y, kv.z, kv.w};
      float va[4] = {vv.x, vv.y, vv.z, vv.w};
#pragma unroll
      for (int cc = 0; cc < 4; ++cc) {
        float pq = qa[cc] > 0.f ? qa[cc] + 1.f : __expf(qa[cc]);
        float pk = ka[cc] > 0.f ? ka[cc] + 1.f : __expf(ka[cc]);
        int d_ = d0 + cc;
        Qb[t * LDQ + d_] = (bf16_t)pq;
        Kb[t * LDQ + d_] = (bf16_t)pk;
        Kts1[d_ * LDT + t] = (bf16_t)(pk * s1);
        Kts2[d_ * LDT + t] = (bf16_t)(pk * s2);
        Vt  [d_ * LDT + t] = (bf16_t)va[cc];
      }
    }
    __syncthreads();

    // ---- phase 2: S = phiQ phiK^T, build scaled causal A ----
#pragma unroll
    for (int it = 0; it < 2; ++it) {
      int id = wave * 2 + it;
      int ti = id >> 2, si = id & 3;
      int trow = ti * 16, scol = si * 16;
      if (si > ti) {  // strictly-upper tile: zero it (wave-uniform branch)
#pragma unroll
        for (int r = 0; r < 8; ++r)
          Alds[(trow + r + 8 * half) * LDT + scol + n] = (bf16_t)0.f;
        continue;
      }
      v8f acc = {};
#pragma unroll
      for (int kd = 0; kd < 4; ++kd) {
        v16bf a  = load_a_bf16(Qb, LDQ, trow, kd * 32, lane);
        v16bf bb = load_b_bf16_t(Kb, LDQ, scol, kd * 32, lane);
        acc = wmma_bf16(a, bb, acc);
      }
      int s_ = scol + n;
      float Ls1 = L1[s_], Ls2 = L2[s_], ms = mvals[s_];
#pragma unroll
      for (int r = 0; r < 8; ++r) {
        int t_ = trow + r + 8 * half;
        float w = (__expf(L1[t_] - Ls1) + __expf(L2[t_] - Ls2)) * ms;
        float av = (s_ <= t_) ? acc[r] * w : 0.f;   // branch-free causal mask
        Alds[t_ * LDT + s_] = (bf16_t)av;
      }
    }
    __syncthreads();

    // ---- phase 2b: rowsum(A) via WMMA (waves 0-3)
    //               phi(q).Z via WMMA against Zb columns (waves 4-7) ----
    if (wave < 4) {
      int trow = wave * 16;
      v8f acc = {};
#pragma unroll
      for (int ks = 0; ks < 2; ++ks) {
        v16bf a  = load_a_bf16(Alds, LDT, trow, ks * 32, lane);
        v16bf bb = load_b_bf16_t(OnesB, LDT, 0, ks * 32, lane);
        acc = wmma_bf16(a, bb, acc);
      }
      if (n == 0) {
#pragma unroll
        for (int r = 0; r < 8; ++r) denA[trow + r + 8 * half] = acc[r];
      }
    } else {
      int trow = (wave - 4) * 16;
      v8f acc = {};
#pragma unroll
      for (int kd = 0; kd < 4; ++kd) {
        v16bf a  = load_a_bf16(Qb, LDQ, trow, kd * 32, lane);
        v16bf bb = load_b_bf16_t(Zb, LDQ, 0, kd * 32, lane);
        acc = wmma_bf16(a, bb, acc);
      }
      if (n == 0) {        // column 0 = phi(q).Z1
#pragma unroll
        for (int r = 0; r < 8; ++r) denC1[trow + r + 8 * half] = acc[r];
      } else if (n == 1) { // column 1 = phi(q).Z2
#pragma unroll
        for (int r = 0; r < 8; ++r) denC2[trow + r + 8 * half] = acc[r];
      }
    }
    __syncthreads();

    if (tid < CHUNK) {
      float den = fmaxf(denA[tid] + E1[tid] * denC1[tid] + E2[tid] * denC2[tid], EPSV);
      float os = mvals[tid] / den;
      oscal[tid] = os;
      O1[tid] = E1[tid] * os;
      O2[tid] = E2[tid] * os;
    }
    __syncthreads();

    // ---- phase 3: O = E1.(Q H1) + E2.(Q H2) + A V ----
#pragma unroll
    for (int it = 0; it < 4; ++it) {
      int id = wave * 4 + it;
      int ti = id >> 3, di = id & 7;
      int trow = ti * 16, dcol = di * 16;
      v8f acc1 = {}, acc2 = {}, accv = {};
#pragma unroll
      for (int kd = 0; kd < 4; ++kd) {
        v16bf aq = load_a_bf16(Qb, LDQ, trow, kd * 32, lane);
        v16bf h1 = load_b_bf16_t(Hb1, LDQ, dcol, kd * 32, lane);
        acc1 = wmma_bf16(aq, h1, acc1);
        v16bf h2 = load_b_bf16_t(Hb2, LDQ, dcol, kd * 32, lane);
        acc2 = wmma_bf16(aq, h2, acc2);
      }
#pragma unroll
      for (int ks = 0; ks < 2; ++ks) {
        v16bf a  = load_a_bf16(Alds, LDT, trow, ks * 32, lane);
        v16bf bb = load_b_bf16_t(Vt, LDT, dcol, ks * 32, lane);
        accv = wmma_bf16(a, bb, accv);
      }
#pragma unroll
      for (int r = 0; r < 8; ++r) {
        int t_ = trow + r + 8 * half;
        float o = acc1[r] * O1[t_] + acc2[r] * O2[t_] + accv[r] * oscal[t_];
        outB[(size_t)(t0 + t_) * DDIM + dcol + n] = o;
      }
    }
    __syncthreads();  // all H reads finished before update

    // ---- phase 4: H <- gamma*H + Khat^T V (f32 master + bf16 shadow) ----
    const float g1 = __expf(LC1), g2 = __expf(LC2);
    {
      const int st = wave >> 2;                 // wave-uniform state select
      float*  Hf  = st ? Hf2 : Hf1;
      bf16_t* Hb  = st ? Hb2 : Hb1;
      bf16_t* Kts = st ? Kts2 : Kts1;
      const float g = st ? g2 : g1;
      const int wq = wave & 3;
#pragma unroll
      for (int it = 0; it < 16; ++it) {
        int rem = wq * 16 + it;                 // 0..63
        int dk0 = (rem >> 3) * 16, dv0 = (rem & 7) * 16;
        v8f acc;
#pragma unroll
        for (int r = 0; r < 8; ++r)
          acc[r] = g * Hf[(dv0 + n) * LDH + dk0 + r + 8 * half];
#pragma unroll
        for (int ks = 0; ks < 2; ++ks) {
          v16bf a  = load_a_bf16(Kts, LDT, dk0, ks * 32, lane);
          v16bf bb = load_b_bf16_t(Vt, LDT, dv0, ks * 32, lane);
          acc = wmma_bf16(a, bb, acc);
        }
#pragma unroll
        for (int r = 0; r < 8; ++r) {
          Hf[(dv0 + n) * LDH + dk0 + r + 8 * half] = acc[r];
          Hb[(dv0 + n) * LDQ + dk0 + r + 8 * half] = (bf16_t)acc[r];
        }
      }
    }
    // ---- Z <- gamma*Z + colsum(Khat) via WMMA (Khat^T . ones) ----
#pragma unroll
    for (int jt = 0; jt < 2; ++jt) {
      int job = wave * 2 + jt;                  // 0..15
      int zst = job >> 3, dk0 = (job & 7) * 16;
      bf16_t* Kts = zst ? Kts2 : Kts1;
      float*  Zf  = zst ? Z2 : Z1;
      float   g   = zst ? g2 : g1;
      v8f acc = {};
#pragma unroll
      for (int ks = 0; ks < 2; ++ks) {
        v16bf a  = load_a_bf16(Kts, LDT, dk0, ks * 32, lane);
        v16bf bb = load_b_bf16_t(OnesB, LDT, 0, ks * 32, lane);
        acc = wmma_bf16(a, bb, acc);
      }
      if (n == 0) {
#pragma unroll
        for (int r = 0; r < 8; ++r) {
          int d_ = dk0 + r + 8 * half;
          float z = g * Zf[d_] + acc[r];
          Zf[d_] = z;
          Zb[zst * LDQ + d_] = (bf16_t)z;       // staged for next chunk's den-cross
        }
      }
    }
    __syncthreads();  // staging buffers safe to overwrite next chunk
  }
}

extern "C" void kernel_launch(void* const* d_in, const int* in_sizes, int n_in,
                              void* d_out, int out_size, void* d_ws, size_t ws_size,
                              hipStream_t stream) {
  (void)n_in; (void)d_ws; (void)ws_size; (void)out_size;
  const float* q    = (const float*)d_in[0];
  const float* k    = (const float*)d_in[1];
  const float* v    = (const float*)d_in[2];
  const float* beta = (const float*)d_in[3];
  const float* mask = (const float*)d_in[4];
  const float* bb1  = (const float*)d_in[5];
  const float* bb2  = (const float*)d_in[6];
  float* out = (float*)d_out;
  const int B = in_sizes[0] / (T_TOTAL * DDIM);  // = 16
  hipLaunchKernelGGL(delta_chunk_kernel, dim3(B), dim3(256), LDS_BYTES, stream,
                     q, k, v, beta, mask, bb1, bb2, out);
}
